// RelativePositionLayer_26182120636453
// MI455X (gfx1250) — compile-verified, matrix operand order checked by reading, not run on hardware
//
#include <hip/hip_runtime.h>
#include <hip/hip_bf16.h>

// Problem constants (from reference setup_inputs)
#define BATCH   4
#define HEADS   8
#define SQ      2048
#define SK      1024
#define DHEAD   64
#define NPOS    258     // valid position ids (row 0 is padding -> zero)
#define PPAD    272     // NPOS padded up to a multiple of 16 (17 N-tiles)
#define NTILES  (PPAD / 16)
#define ROWS    16      // q rows per workgroup tile

typedef __attribute__((ext_vector_type(16))) _Float16 v16h;
typedef __attribute__((ext_vector_type(8)))  float    v8f;
typedef __attribute__((ext_vector_type(4)))  float    v4f;
typedef __attribute__((ext_vector_type(4)))  int      v4i;

// ---------------------------------------------------------------------------
// Prep: build per-head embedding in f16 into workspace.
//   E[h][p][k] = (1 <= p < NPOS) ? table[p*512 + k*8 + h] : 0   (f16)
// Layout: h*PPAD*64 + p*64 + k  (rows 128B contiguous; whole thing 272KB,
// stays hot in L2 for the main kernel's B-fragment loads).
// ---------------------------------------------------------------------------
__global__ void __launch_bounds__(256)
relpos_prep_E(const float* __restrict__ table, _Float16* __restrict__ E) {
    int i = blockIdx.x * 256 + threadIdx.x;        // 0 .. HEADS*PPAD*64-1
    int k  = i & 63;
    int hp = i >> 6;
    int p  = hp % PPAD;
    int h  = hp / PPAD;
    float v = 0.0f;
    if (p >= 1 && p < NPOS) v = table[p * 512 + k * 8 + h];
    E[i] = (_Float16)v;
}

// ---------------------------------------------------------------------------
// Fused kernel: one workgroup = (batch b, 16-row q tile), ALL 8 heads.
//   Phase 1 (WMMA): wave w computes head h=w:
//       Qrel[h][m][p] = sum_k q[b,h,qbase+m,k] * E[h][p][k]   -> LDS
//   Phase 2 (gather): out[b*8+h, qbase+r, k] = LDS[h][r][ idx[b,qbase+r,k] ]
// Dynamic LDS: HEADS*ROWS*PPAD*4 = 139,264 bytes.
// ---------------------------------------------------------------------------
__global__ void __launch_bounds__(256)
relpos_main(const float* __restrict__ q,
            const _Float16* __restrict__ E,
            const int* __restrict__ idx,
            float* __restrict__ out) {
    extern __shared__ float lds[];                 // [HEADS][ROWS][PPAD]

    const int qtile = blockIdx.x;                  // 0..127
    const int b     = blockIdx.y;                  // 0..3
    const int qbase = qtile * ROWS;
    const int tid   = threadIdx.x;
    const int wave  = tid >> 5;                    // 0..7 == head for phase 1
    const int lane  = tid & 31;

    // ------------------ Phase 1: WMMA GEMM into LDS ------------------
    {
        const int h     = wave;
        const int mn    = lane & 15;               // A: row m   B: col n
        const int khalf = (lane >> 4) * 8;         // K sub-block select

        // A fragments (16x64 f16, K split into two 16x32 WMMA operands).
        // 16-bit A layout (ISA 7.12.2): lane -> M, VGPR v packs K pair
        //   k = (v/4)*16 + (lane/16)*8 + (v%4)*2
        const float* qrow =
            q + (((size_t)(b * HEADS + h)) * SQ + (size_t)(qbase + mn)) * DHEAD;
        v16h a0, a1;
#pragma unroll
        for (int v = 0; v < 8; ++v) {
            int k = (v >> 2) * 16 + khalf + (v & 3) * 2;
            a0[2 * v]     = (_Float16)qrow[k];
            a0[2 * v + 1] = (_Float16)qrow[k + 1];
            a1[2 * v]     = (_Float16)qrow[k + 32];
            a1[2 * v + 1] = (_Float16)qrow[k + 33];
        }

        const int mh = (lane >> 4) * 8;            // C/D: M = r + 8*(lane/16)

#pragma unroll 1
        for (int t = 0; t < NTILES; ++t) {
            // B fragment: B[k][n] = E[h][t*16+n][k]; lane -> n, same K packing.
            const _Float16* erow =
                E + ((size_t)h * PPAD + (size_t)(t * 16 + mn)) * DHEAD;
            v16h b0, b1;
#pragma unroll
            for (int v = 0; v < 8; ++v) {
                int k = (v >> 2) * 16 + khalf + (v & 3) * 2;
                b0[2 * v]     = erow[k];
                b0[2 * v + 1] = erow[k + 1];
                b1[2 * v]     = erow[k + 32];
                b1[2 * v + 1] = erow[k + 33];
            }

            v8f acc = {};
            acc = __builtin_amdgcn_wmma_f32_16x16x32_f16(
                false, a0, false, b0, (short)0, acc, false, false);
            acc = __builtin_amdgcn_wmma_f32_16x16x32_f16(
                false, a1, false, b1, (short)0, acc, false, false);

            // Scatter D tile to LDS: element (m = r + mh, n = mn).
#pragma unroll
            for (int r = 0; r < 8; ++r) {
                lds[(h * ROWS + (r + mh)) * PPAD + t * 16 + mn] = acc[r];
            }
        }
    }

    __syncthreads();

    // ------------------ Phase 2: gather + streamed output ------------------
    // 256 threads x 4 consecutive k -> one full Sk row per pass; each index
    // is read once and reused for all 8 heads.
    const int k0 = tid * 4;
#pragma unroll 1
    for (int r = 0; r < ROWS; ++r) {
        const int* ip = idx + ((size_t)b * SQ + (size_t)(qbase + r)) * SK + k0;
        if (r + 1 < ROWS)
            __builtin_prefetch(ip + SK, 0, 0);     // global_prefetch next row
        v4i ii = __builtin_nontemporal_load((const v4i*)ip);

#pragma unroll
        for (int h = 0; h < HEADS; ++h) {
            const float* row = lds + (h * ROWS + r) * PPAD;
            v4f vals;
            vals.x = row[ii.x];
            vals.y = row[ii.y];
            vals.z = row[ii.z];
            vals.w = row[ii.w];
            float* op = out +
                (((size_t)(b * HEADS + h)) * SQ + (size_t)(qbase + r)) * SK + k0;
            __builtin_nontemporal_store(vals, (v4f*)op);
        }
    }
}

// ---------------------------------------------------------------------------
// Launch. d_ws usage: HEADS*PPAD*DHEAD*2 = 278,528 bytes (f16 embedding).
// ---------------------------------------------------------------------------
extern "C" void kernel_launch(void* const* d_in, const int* in_sizes, int n_in,
                              void* d_out, int out_size, void* d_ws, size_t ws_size,
                              hipStream_t stream) {
    const float* query_proj = (const float*)d_in[0];   // (B*H, Sq, 64) f32
    const float* rel_table  = (const float*)d_in[1];   // (258, 512) f32
    const int*   rel_ids    = (const int*)d_in[2];     // (B, Sq, Sk) int
    float*       out        = (float*)d_out;           // (B*H, Sq, Sk) f32
    _Float16*    E          = (_Float16*)d_ws;

    (void)in_sizes; (void)n_in; (void)out_size; (void)ws_size;

    // Prep: HEADS*PPAD*64 = 139,264 elements -> 544 blocks of 256.
    relpos_prep_E<<<dim3((HEADS * PPAD * DHEAD) / 256), dim3(256), 0, stream>>>(
        rel_table, E);

    const size_t lds_bytes = (size_t)HEADS * ROWS * PPAD * sizeof(float);
    relpos_main<<<dim3(SQ / ROWS, BATCH), dim3(256), lds_bytes, stream>>>(
        query_proj, E, rel_ids, out);
}
